// _CueChannelwiseEdgeConv_13469017440867
// MI455X (gfx1250) — matrix operand order repaired and checked by reading, not compiled, not run on hardware
//
#include <hip/hip_runtime.h>
#include <math.h>

// ---------------------------------------------------------------------------
// CueChannelwiseEdgeConv for MI455X (gfx1250, wave32, WMMA)
//
//   K1 node_up  : xs = s @ Wup0 / sqrt(32)           (N,32)
//                 xv = v @ Wup1 / sqrt(32), planar    (N,3,32)
//   memset      : msg_s (N,64), msg_v (N,3,64) = 0
//   K2 edge     : per-wave 16-edge tile; radial-embedding MLP
//                 (8->64->64->128) via v_wmma_f32_16x16x32_f16; f16 LDS
//                 activations so every WMMA fragment is two ds_load_b128;
//                 async-to-LDS gather of sender xs rows overlapped with the
//                 GEMMs; tensor-product messages, f32 atomic scatter (L2).
//   K3 node_out : out_s = msg_s @ Wout0 * sc ; out_v = msg_v @ Wout1 * sc
// ---------------------------------------------------------------------------

typedef _Float16 h16;
typedef __attribute__((ext_vector_type(16))) _Float16 v16h;
typedef __attribute__((ext_vector_type(8)))  float    v8f;
typedef __attribute__((ext_vector_type(4)))  unsigned int v4u;
typedef __attribute__((ext_vector_type(8)))  unsigned int v8u;
typedef int v4i __attribute__((vector_size(16)));

// gfx1250 async global->LDS copy (ASYNCcnt path); confirmed present on this
// toolchain (signature: (v4i* gsrc, v4i* ldsdst, int offset, int cpol)).
#if defined(__has_builtin)
#if __has_builtin(__builtin_amdgcn_global_load_async_to_lds_b128)
#define HAVE_ASYNC_LDS 1
#endif
#endif
#ifndef HAVE_ASYNC_LDS
#define HAVE_ASYNC_LDS 0
#endif

#if HAVE_ASYNC_LDS
__device__ __forceinline__ void wait_async0() {
#if __has_builtin(__builtin_amdgcn_s_wait_asynccnt)
  __builtin_amdgcn_s_wait_asynccnt(0);
#else
  asm volatile("s_wait_asynccnt 0" ::: "memory");
#endif
}
#endif

__device__ __forceinline__ v8f wmma16(v16h a, v16h b, v8f c) {
  // D = A(16x32 f16) * B(32x16 f16) + C(16x16 f32)
  return __builtin_amdgcn_wmma_f32_16x16x32_f16(
      /*neg_a=*/false, a, /*neg_b=*/false, b,
      /*c_mod=*/(short)0, c, /*reuse_a=*/false, /*reuse_b=*/false);
}

// Fast silu: v_exp + v_rcp instead of the IEEE division macro.
__device__ __forceinline__ float silu(float x) {
  return x * __builtin_amdgcn_rcpf(1.0f + __expf(-x));
}

// A-tile loader from an f16 LDS activation block laid out [m][64].
// A k order used here, contiguous (matches load_b16): lanes 0-15 (lh=0): K0..15
//                                           lanes 16-31(lh=1): K16..31
// Both 8-element runs are contiguous in LDS -> two ds_load_b128, no cvts.
__device__ __forceinline__ v16h load_a16(const h16* __restrict__ h, int m,
                                         int lh, int koff) {
  const h16* base = h + m * 64 + koff + 16 * lh;
  v4u lo = *(const v4u*)(base);       // K = koff+16lh   .. +7
  v4u hi = *(const v4u*)(base + 8);   // K = koff+16lh+8 .. +7
  v8u u;
  u[0] = lo[0]; u[1] = lo[1]; u[2] = lo[2]; u[3] = lo[3];
  u[4] = hi[0]; u[5] = hi[1]; u[6] = hi[2]; u[7] = hi[3];
  return __builtin_bit_cast(v16h, u);
}

// B-tile loader from LDS f16 weights stored transposed [n][ldk] (k contig).
// B k order used here, contiguous: lanes 0-15 hold K=0..15, lanes 16-31 K=16..31.
// The 16 halves a lane needs are one contiguous 32B run -> two ds_load_b128.
__device__ __forceinline__ v16h load_b16(const h16* __restrict__ wT, int n,
                                         int lh, int koff, int ldk) {
  v8u u = *(const v8u*)(wT + n * ldk + koff + 16 * lh);
  return __builtin_bit_cast(v16h, u);
}

// ---------------------------------------------------------------------------
__global__ __launch_bounds__(128) void node_up_kernel(
    const float* __restrict__ nf, const float* __restrict__ Wu0,
    const float* __restrict__ Wu1, float* __restrict__ xs,
    float* __restrict__ xv, int N) {
  int t = blockIdx.x * blockDim.x + threadIdx.x;
  int n = t >> 5, k = t & 31;
  if (n >= N) return;
  const float* row = nf + (size_t)n * 128;
  float as = 0.f, a0 = 0.f, a1 = 0.f, a2 = 0.f;
#pragma unroll 8
  for (int m = 0; m < 32; ++m) {
    float u0 = Wu0[m * 32 + k], u1 = Wu1[m * 32 + k];
    as += row[m] * u0;
    a0 += row[32 + m * 3 + 0] * u1;
    a1 += row[32 + m * 3 + 1] * u1;
    a2 += row[32 + m * 3 + 2] * u1;
  }
  const float sc = 0.17677669529663687f;  // 1/sqrt(32)
  xs[(size_t)n * 32 + k] = as * sc;
  xv[((size_t)n * 3 + 0) * 32 + k] = a0 * sc;
  xv[((size_t)n * 3 + 1) * 32 + k] = a1 * sc;
  xv[((size_t)n * 3 + 2) * 32 + k] = a2 * sc;
}

// ---------------------------------------------------------------------------
__global__ __launch_bounds__(128) void edge_kernel(
    const float* __restrict__ edge_vec, const float* __restrict__ edge_len,
    const int* __restrict__ senders, const int* __restrict__ receivers,
    const float* __restrict__ w0, const float* __restrict__ b0,
    const float* __restrict__ w1, const float* __restrict__ b1,
    const float* __restrict__ w2, const float* __restrict__ b2,
    const float* __restrict__ xs, const float* __restrict__ xv,
    float* __restrict__ msg_s, float* __restrict__ msg_v, int E) {
  __shared__ __align__(16) h16 w0T[64 * 8];       //  1 KB  mlp_w0^T (f16)
  __shared__ __align__(16) h16 w1T[64 * 64];      //  8 KB  mlp_w1^T (f16)
  __shared__ __align__(16) h16 w2T[128 * 64];     // 16 KB  mlp_w2^T (f16)
  __shared__ __align__(16) h16 hbuf[4][16 * 64];  //  8 KB  acts (h1 then h2)
  __shared__ __align__(16) float gatherS[4][16][32];  // 8 KB async xs rows
  __shared__ float metaY[4][16][3];
  __shared__ int   metaS[4][16];
  __shared__ int   metaR[4][16];

  const int tid = threadIdx.x;
  const int wave = tid >> 5;
  const int lane = tid & 31;
  const int lh = lane >> 4;
  const int n0 = lane & 15;

  // Stage MLP weights into LDS, f32 -> f16, transposed to [n][k].
  for (int i = tid; i < 64 * 8; i += 128) {
    int n = i >> 3, k = i & 7;
    w0T[i] = (h16)w0[k * 64 + n];
  }
  for (int i = tid; i < 64 * 64; i += 128) {
    int n = i >> 6, k = i & 63;
    w1T[i] = (h16)w1[k * 64 + n];
  }
  for (int i = tid; i < 128 * 64; i += 128) {
    int n = i >> 6, k = i & 63;
    w2T[i] = (h16)w2[k * 128 + n];
  }
  __syncthreads();

  // -------- per-edge scalar prep (lanes 0..15 own one edge each) -----------
  float emb[8];
#pragma unroll
  for (int j = 0; j < 8; ++j) emb[j] = 0.f;
  const int ebase = blockIdx.x * 64 + wave * 16;
  if (lane < 16) {
    int e = ebase + lane;
    if (e < E) {
      float vx = edge_vec[e * 3 + 0];
      float vy = edge_vec[e * 3 + 1];
      float vz = edge_vec[e * 3 + 2];
      float inv = 1.7320508075688772f /
                  (sqrtf(vx * vx + vy * vy + vz * vz) + 1e-12f);  // sqrt(3)/|v|
      metaY[wave][lane][0] = vx * inv;
      metaY[wave][lane][1] = vy * inv;
      metaY[wave][lane][2] = vz * inv;
      metaS[wave][lane] = senders[e];
      metaR[wave][lane] = receivers[e];
      float r = edge_len[e];
      float x = fminf(fmaxf(r, 0.f), 1.f);
      float gate = (r <= 1.0f) ? 2.8284271247461903f : 0.f;  // sqrt(8)
#pragma unroll
      for (int j = 0; j < 8; ++j) {
        float t = (x - (float)j * (1.f / 7.f)) * 7.f;
        emb[j] = gate * __expf(-0.5f * t * t);
      }
    } else {
      metaR[wave][lane] = -1;
      metaS[wave][lane] = 0;
      metaY[wave][lane][0] = metaY[wave][lane][1] = metaY[wave][lane][2] = 0.f;
    }
  }

#if HAVE_ASYNC_LDS
  // Kick off async gather of the 16 sender xs rows (32 f32 each) into LDS.
  // Each lane moves 64B: lane = (half lh of row n0). Overlaps with the GEMMs.
  {
    int sn = metaS[wave][n0];  // in-order DS: prep stores are visible
    const float* src = xs + (size_t)sn * 32 + lh * 16;
    float* dst = &gatherS[wave][n0][lh * 16];
#pragma unroll
    for (int j = 0; j < 4; ++j) {
      __builtin_amdgcn_global_load_async_to_lds_b128(
          (v4i*)(src + j * 4), (v4i*)(dst + j * 4), 0, 0);
    }
  }
#endif

  // -------- GEMM1: (16x8 pad 32) x (8x64)  -> h1 ---------------------------
  v16h a1;
#pragma unroll
  for (int i = 0; i < 16; ++i)
    a1[i] = (lh == 0 && i < 8) ? (h16)emb[i] : (h16)0.f;

  h16* hw = hbuf[wave];
  const bool blo = (lh == 0);
#pragma unroll
  for (int nt = 0; nt < 4; ++nt) {
    int nc = n0 + 16 * nt;
    // Unconditional contiguous 16B LDS load + cndmask (no exec branching).
    v4u w8 = *(const v4u*)(w0T + nc * 8);  // halves K=0..7 of column nc
    v8u ub;
#pragma unroll
    for (int j = 0; j < 4; ++j) ub[j] = blo ? w8[j] : 0u;
#pragma unroll
    for (int j = 4; j < 8; ++j) ub[j] = 0u;
    v16h bt = __builtin_bit_cast(v16h, ub);
    v8f c;
    float bias = b0[nc];
#pragma unroll
    for (int rr = 0; rr < 8; ++rr) c[rr] = bias;
    c = wmma16(a1, bt, c);
#pragma unroll
    for (int rr = 0; rr < 8; ++rr)
      hw[(rr + 8 * lh) * 64 + nc] = (h16)silu(c[rr]);
  }

  // -------- GEMM2: (16x64) x (64x64); accumulators stay in registers -------
  v8f c2[4];
#pragma unroll
  for (int nt = 0; nt < 4; ++nt) {
    int nc = n0 + 16 * nt;
    v8f c;
    float bias = b1[nc];
#pragma unroll
    for (int rr = 0; rr < 8; ++rr) c[rr] = bias;
#pragma unroll
    for (int ks = 0; ks < 2; ++ks)
      c = wmma16(load_a16(hw, n0, lh, 32 * ks),
                 load_b16(w1T, nc, lh, 32 * ks, 64), c);
    c2[nt] = c;
  }
  // All h1 reads are done (per-wave DS is in-order); overwrite hbuf with h2.
#pragma unroll
  for (int nt = 0; nt < 4; ++nt) {
    int nc = n0 + 16 * nt;
#pragma unroll
    for (int rr = 0; rr < 8; ++rr)
      hw[(rr + 8 * lh) * 64 + nc] = (h16)silu(c2[nt][rr]);
  }

  // -------- GEMM3: (16x64) x (64x128) -> w (kept in C registers) -----------
  v8f cw[8];
#pragma unroll
  for (int nt = 0; nt < 8; ++nt) {
    int nc = n0 + 16 * nt;
    v8f c;
    float bias = b2[nc];
#pragma unroll
    for (int rr = 0; rr < 8; ++rr) c[rr] = bias;
#pragma unroll
    for (int ks = 0; ks < 2; ++ks)
      c = wmma16(load_a16(hw, n0, lh, 32 * ks),
                 load_b16(w2T, nc, lh, 32 * ks, 64), c);
    cw[nt] = c;
  }

#if HAVE_ASYNC_LDS
  wait_async0();  // gathered xs rows now valid in LDS
#endif

  // -------- message phase directly from C-register layout ------------------
  // Lane holds w[e][n0 + 16*nt] for edges e = rr + 8*lh. For channel
  // c = n0 + 16*cc, the four w-blocks (cols c, 32+c, 64+c, 96+c) live in
  // tiles nt = cc, cc+2, cc+4, cc+6.  => 2 channels x 8 edges per lane.
  const float inv_sq3 = 0.5773502691896258f;
#pragma unroll
  for (int rr = 0; rr < 8; ++rr) {
    int el = rr + 8 * lh;
    int rc = metaR[wave][el];
    if (rc < 0) continue;
    int sn = metaS[wave][el];
    float y0 = metaY[wave][el][0];
    float y1 = metaY[wave][el][1];
    float y2 = metaY[wave][el][2];
#if HAVE_ASYNC_LDS
    const float* xsn = &gatherS[wave][el][0];
    (void)sn;
#else
    const float* xsn = xs + (size_t)sn * 32;
#endif
    const float* xvn = xv + (size_t)sn * 96;
    float* msS = msg_s + (size_t)rc * 64;
    float* msV = msg_v + (size_t)rc * 192;
#pragma unroll
    for (int cc = 0; cc < 2; ++cc) {
      int ch = n0 + 16 * cc;
      float wm0 = cw[cc + 0][rr];
      float wm1 = cw[cc + 2][rr];
      float wm2 = cw[cc + 4][rr];
      float wm3 = cw[cc + 6][rr];
      float ss = xsn[ch];
      float v0 = xvn[0 * 32 + ch];
      float v1 = xvn[1 * 32 + ch];
      float v2 = xvn[2 * 32 + ch];
      float dt = (v0 * y0 + v1 * y1 + v2 * y2) * inv_sq3;
      atomicAdd(&msS[ch], wm0 * ss);            // m0
      atomicAdd(&msS[32 + ch], wm3 * dt);       // m3
      float a1s = wm1 * ss;                     // m1 = w1*ss*Y1
      atomicAdd(&msV[0 * 64 + ch], a1s * y0);
      atomicAdd(&msV[1 * 64 + ch], a1s * y1);
      atomicAdd(&msV[2 * 64 + ch], a1s * y2);
      atomicAdd(&msV[0 * 64 + 32 + ch], wm2 * v0);  // m2 = w2*vv
      atomicAdd(&msV[1 * 64 + 32 + ch], wm2 * v1);
      atomicAdd(&msV[2 * 64 + 32 + ch], wm2 * v2);
    }
  }
}

// ---------------------------------------------------------------------------
__global__ __launch_bounds__(128) void node_out_kernel(
    const float* __restrict__ msg_s, const float* __restrict__ msg_v,
    const float* __restrict__ Wo0, const float* __restrict__ Wo1,
    float* __restrict__ out, int N) {
  int t = blockIdx.x * blockDim.x + threadIdx.x;
  int n = t >> 5, k = t & 31;
  if (n >= N) return;
  const float* ms = msg_s + (size_t)n * 64;
  const float* mv = msg_v + (size_t)n * 192;
  float as = 0.f, a0 = 0.f, a1 = 0.f, a2 = 0.f;
#pragma unroll 8
  for (int j = 0; j < 64; ++j) {
    float o0 = Wo0[j * 32 + k], o1 = Wo1[j * 32 + k];
    as += ms[j] * o0;
    a0 += mv[j] * o1;
    a1 += mv[64 + j] * o1;
    a2 += mv[128 + j] * o1;
  }
  const float sc = 0.0078125f;  // (1/sqrt(64)) / AVG(16)
  float* orow = out + (size_t)n * 128;
  orow[k] = as * sc;
  orow[32 + k * 3 + 0] = a0 * sc;
  orow[32 + k * 3 + 1] = a1 * sc;
  orow[32 + k * 3 + 2] = a2 * sc;
}

// ---------------------------------------------------------------------------
extern "C" void kernel_launch(void* const* d_in, const int* in_sizes, int n_in,
                              void* d_out, int out_size, void* d_ws,
                              size_t ws_size, hipStream_t stream) {
  const float* node_feats = (const float*)d_in[0];
  const float* edge_vec   = (const float*)d_in[1];
  const float* edge_len   = (const float*)d_in[2];
  const int*   senders    = (const int*)d_in[3];
  const int*   receivers  = (const int*)d_in[4];
  // d_in[5] = num_nodes scalar (derive N from sizes instead)
  const float* W_up0 = (const float*)d_in[6];
  const float* W_up1 = (const float*)d_in[7];
  const float* w0 = (const float*)d_in[8];
  const float* b0 = (const float*)d_in[9];
  const float* w1 = (const float*)d_in[10];
  const float* b1 = (const float*)d_in[11];
  const float* w2 = (const float*)d_in[12];
  const float* b2 = (const float*)d_in[13];
  const float* Wo0 = (const float*)d_in[14];
  const float* Wo1 = (const float*)d_in[15];

  const int N = in_sizes[0] / 128;
  const int E = in_sizes[1] / 3;

  // Workspace: xs (N*32) | xv (N*96) | msg_s (N*64) | msg_v (N*192)
  float* xs = (float*)d_ws;
  float* xv = xs + (size_t)N * 32;
  float* msg_s = xv + (size_t)N * 96;
  float* msg_v = msg_s + (size_t)N * 64;

  (void)hipMemsetAsync(msg_s, 0, (size_t)N * 256 * sizeof(float), stream);

  int nodeBlocks = (N * 32 + 127) / 128;
  node_up_kernel<<<nodeBlocks, 128, 0, stream>>>(node_feats, W_up0, W_up1, xs,
                                                 xv, N);

  int edgeBlocks = (E + 63) / 64;
  edge_kernel<<<edgeBlocks, 128, 0, stream>>>(edge_vec, edge_len, senders,
                                              receivers, w0, b0, w1, b1, w2, b2,
                                              xs, xv, msg_s, msg_v, E);

  node_out_kernel<<<nodeBlocks, 128, 0, stream>>>(msg_s, msg_v, Wo0, Wo1,
                                                  (float*)d_out, N);
}
